// PointNetPolylineEncoder_11922829214230
// MI455X (gfx1250) — compile-verified, matrix operand order checked by reading, not run on hardware
//
#include <hip/hip_runtime.h>

typedef __bf16 bf16;
typedef __attribute__((ext_vector_type(16))) __bf16 v16bf;
typedef __attribute__((ext_vector_type(8)))  float  v8f;

union FragBF {
  v16bf v;
  uint4 q[2];
};

// ---------------------------------------------------------------------------
// WMMA GEMM: D[M,N] = A[M,K] * W[N,K]^T   (bf16 inputs, fp32 accumulate)
// One wave computes a 64(M) x 64(N) tile: 4 A-frags x 4 B-frags = 16
// independent WMMAs per K-step (hides cache latency, 16 B/lane/WMMA).
// EPI 0: store fp32 raw.  EPI 1: v=relu(v+bias)->bf16.  EPI 2: v=(v+bias)*valid[row]->fp32.
// CONCAT: A row = [ X[row, 0:256] | Pool[row/20, 0:256] ]  (lda applies to X part)
// ---------------------------------------------------------------------------
template<int EPI, bool CONCAT>
__global__ __launch_bounds__(256) void gemm_wmma(
    const bf16* __restrict__ A, const bf16* __restrict__ Apool,
    const bf16* __restrict__ W, const float* __restrict__ bias,
    const float* __restrict__ valid, float* __restrict__ outF,
    bf16* __restrict__ outB, int M, int N, int K, int lda)
{
  const int wave = (blockIdx.x * 256 + threadIdx.x) >> 5;
  const int lane = threadIdx.x & 31;
  const int nGroups = N >> 6;                // N/64
  const int tileM = (wave / nGroups) << 6;   // 64-row tile
  const int nBase = (wave % nGroups) << 6;   // 64-col tile
  if (tileM >= M) return;                    // whole-wave uniform

  const int half = lane >> 4;                // 0: K {0..7,16..23}, 1: K {8..15,24..31}
  const int l15  = lane & 15;
  const int off  = half ? 8 : 0;

  int rowA[4], prow[4];
#pragma unroll
  for (int i = 0; i < 4; ++i) {
    rowA[i] = tileM + i * 16 + l15;
    prow[i] = CONCAT ? (rowA[i] / 20) : 0;
  }

  v8f acc[4][4] = {};

  for (int k0 = 0; k0 < K; k0 += 32) {
    const int kk0 = k0 + off;
    const int kk1 = kk0 + 16;

    FragBF a[4];
#pragma unroll
    for (int i = 0; i < 4; ++i) {
      if (CONCAT) {
        const bf16* p0 = (kk0 < 256) ? (A + (size_t)rowA[i] * lda + kk0)
                                     : (Apool + (size_t)prow[i] * 256 + (kk0 - 256));
        const bf16* p1 = (kk1 < 256) ? (A + (size_t)rowA[i] * lda + kk1)
                                     : (Apool + (size_t)prow[i] * 256 + (kk1 - 256));
        a[i].q[0] = *(const uint4*)p0;
        a[i].q[1] = *(const uint4*)p1;
      } else {
        const bf16* p = A + (size_t)rowA[i] * lda + kk0;
        a[i].q[0] = *(const uint4*)p;
        a[i].q[1] = *(const uint4*)(p + 16);
      }
    }

    FragBF b[4];
#pragma unroll
    for (int j = 0; j < 4; ++j) {
      const int n = nBase + j * 16 + l15;
      const bf16* pw = W + (size_t)n * K + kk0;
      b[j].q[0] = *(const uint4*)pw;
      b[j].q[1] = *(const uint4*)(pw + 16);
    }

#pragma unroll
    for (int i = 0; i < 4; ++i)
#pragma unroll
      for (int j = 0; j < 4; ++j)
        acc[i][j] = __builtin_amdgcn_wmma_f32_16x16x32_bf16(
            false, a[i].v, false, b[j].v, (short)0, acc[i][j], false, false);
  }

#pragma unroll
  for (int i = 0; i < 4; ++i) {
    const int rBase = tileM + i * 16 + (half ? 8 : 0);
#pragma unroll
    for (int j = 0; j < 4; ++j) {
      const int gn = nBase + j * 16 + l15;
#pragma unroll
      for (int r = 0; r < 8; ++r) {
        const int gm = rBase + r;
        const size_t idx = (size_t)gm * N + gn;
        float v = acc[i][j][r];
        if (EPI == 0) {
          outF[idx] = v;
        } else if (EPI == 1) {
          v = fmaxf(v + bias[gn], 0.f);
          outB[idx] = (bf16)v;
        } else {
          v = (v + bias[gn]) * valid[gm];
          outF[idx] = v;
        }
      }
    }
  }
}

// ---------------------------------------------------------------------------
// Masked per-channel sum / sum-of-squares (256 channels, one per thread)
// ---------------------------------------------------------------------------
__global__ __launch_bounds__(256) void bn_stats(
    const float* __restrict__ act, const float* __restrict__ mrow,
    float* __restrict__ sum, float* __restrict__ sq, int rows)
{
  const int c = threadIdx.x;
  const int rowsPer = (rows + gridDim.x - 1) / gridDim.x;
  const int r0 = blockIdx.x * rowsPer;
  int r1 = r0 + rowsPer; if (r1 > rows) r1 = rows;
  float s = 0.f, ss = 0.f;
  for (int r = r0; r < r1; ++r) {
    const float m = mrow[r];
    const float v = act[(size_t)r * 256 + c];
    s  += v * m;
    ss += v * v * m;
  }
  atomicAdd(&sum[c], s);
  atomicAdd(&sq[c], ss);
}

__global__ __launch_bounds__(256) void bn_finalize(
    const float* __restrict__ sum, const float* __restrict__ sq,
    const float* __restrict__ g, const float* __restrict__ b,
    const float* __restrict__ cnt, float* __restrict__ scale,
    float* __restrict__ shift)
{
  const int c = threadIdx.x;
  const float n = fmaxf(*cnt, 1.0f);
  const float mean = sum[c] / n;
  const float var  = sq[c] / n - mean * mean;
  const float sc   = g[c] * rsqrtf(var + 1e-5f);
  scale[c] = sc;
  shift[c] = b[c] - mean * sc;
}

// MODE 0: write bf16 activations + bf16 pool.  MODE 1: activations only.
// MODE 2: pool only (buf).
template<int MODE>
__global__ __launch_bounds__(256) void bn_apply(
    const float* __restrict__ act, const float* __restrict__ mrow,
    const float* __restrict__ scale, const float* __restrict__ shift,
    bf16* __restrict__ actOut, bf16* __restrict__ poolOut)
{
  const int p = blockIdx.x;      // polyline
  const int c = threadIdx.x;     // channel
  const float sc = scale[c], sh = shift[c];
  float mx = 0.f;
#pragma unroll
  for (int n = 0; n < 20; ++n) {
    const size_t r = (size_t)p * 20 + n;
    float v = fmaxf(act[r * 256 + c] * sc + sh, 0.f) * mrow[r];
    if (MODE != 2) actOut[r * 256 + c] = (bf16)v;
    mx = fmaxf(mx, v);
  }
  if (MODE != 1) poolOut[(size_t)p * 256 + c] = (bf16)mx;
}

// mask -> float per point row; block-reduced count into *cnt (atomic)
__global__ __launch_bounds__(256) void mask_kernel(
    const int* __restrict__ mask, float* __restrict__ mrow,
    float* __restrict__ cnt, int n)
{
  __shared__ float red[256];
  const int i = blockIdx.x * 256 + threadIdx.x;
  float m = 0.f;
  if (i < n) { m = (mask[i] != 0) ? 1.f : 0.f; mrow[i] = m; }
  red[threadIdx.x] = m;
  __syncthreads();
  for (int s = 128; s > 0; s >>= 1) {
    if (threadIdx.x < s) red[threadIdx.x] += red[threadIdx.x + s];
    __syncthreads();
  }
  if (threadIdx.x == 0) atomicAdd(cnt, red[0]);
}

__global__ __launch_bounds__(256) void valid_kernel(
    const int* __restrict__ mask, float* __restrict__ valid, int P)
{
  const int p = blockIdx.x * 256 + threadIdx.x;
  if (p >= P) return;
  int any = 0;
#pragma unroll
  for (int n = 0; n < 20; ++n) any |= mask[p * 20 + n];
  valid[p] = any ? 1.f : 0.f;
}

__global__ __launch_bounds__(256) void cvt_bf16(
    const float* __restrict__ src, bf16* __restrict__ dst, int n)
{
  const int i = blockIdx.x * 256 + threadIdx.x;
  if (i < n) dst[i] = (bf16)src[i];
}

// fp32 [rows,kin] -> bf16 [rows,kout], zero-padded
__global__ __launch_bounds__(256) void pad_bf16(
    const float* __restrict__ src, bf16* __restrict__ dst,
    int rows, int kin, int kout)
{
  const int i = blockIdx.x * 256 + threadIdx.x;
  if (i >= rows * kout) return;
  const int k = i % kout, r = i / kout;
  dst[i] = (k < kin) ? (bf16)src[(size_t)r * kin + k] : (bf16)0.f;
}

// ---------------------------------------------------------------------------
extern "C" void kernel_launch(void* const* d_in, const int* in_sizes, int n_in,
                              void* d_out, int out_size, void* d_ws, size_t ws_size,
                              hipStream_t stream) {
  (void)in_sizes; (void)n_in; (void)out_size; (void)ws_size;
  constexpr int Bn = 16, P = 768, Np = 20, C = 9, H = 256;
  constexpr int ROWS = Bn * P * Np;    // 245760
  constexpr int POLY = Bn * P;         // 12288

  const float* polylines = (const float*)d_in[0];
  const int*   mask      = (const int*)d_in[1];
  const float* W_pre     = (const float*)d_in[2];
  const float* g_pre     = (const float*)d_in[3];
  const float* b_pre     = (const float*)d_in[4];
  const float* W1        = (const float*)d_in[5];
  const float* g1        = (const float*)d_in[6];
  const float* b1        = (const float*)d_in[7];
  const float* W2        = (const float*)d_in[8];
  const float* g2        = (const float*)d_in[9];
  const float* b2        = (const float*)d_in[10];
  const float* Wo1       = (const float*)d_in[11];
  const float* bo1       = (const float*)d_in[12];
  const float* Wo2       = (const float*)d_in[13];
  const float* bo2       = (const float*)d_in[14];
  float* out = (float*)d_out;

  char* ws = (char*)d_ws;
  // --- stats region (first 8192 bytes zeroed each call) ---
  float* statsZ = (float*)ws;
  float* sum1 = statsZ + 0,    *sq1 = statsZ + 256;
  float* sum2 = statsZ + 512,  *sq2 = statsZ + 768;
  float* sum3 = statsZ + 1024, *sq3 = statsZ + 1280;
  float* cnt  = statsZ + 1536;
  float* scale1 = statsZ + 2048, *shift1 = statsZ + 2304;
  float* scale2 = statsZ + 2560, *shift2 = statsZ + 2816;
  float* scale3 = statsZ + 3072, *shift3 = statsZ + 3328;

  size_t o = 16384;
  auto alloc = [&](size_t bytes) -> char* {
    char* p = ws + o;
    o = (o + bytes + 255) & ~(size_t)255;
    return p;
  };
  float* actF    = (float*)alloc((size_t)ROWS * H * 4);   // raw GEMM outputs
  bf16*  actB    = (bf16*) alloc((size_t)ROWS * H * 2);   // bf16 activations
  bf16*  padA    = (bf16*) alloc((size_t)ROWS * 32 * 2);  // padded bf16 input
  bf16*  pooledB = (bf16*) alloc((size_t)POLY * H * 2);
  bf16*  bufB    = (bf16*) alloc((size_t)POLY * H * 2);
  bf16*  o1B     = (bf16*) alloc((size_t)POLY * H * 2);
  float* mrow    = (float*)alloc((size_t)ROWS * 4);
  float* validF  = (float*)alloc((size_t)POLY * 4);
  bf16*  WpreB   = (bf16*) alloc((size_t)H * 32 * 2);
  bf16*  W1B     = (bf16*) alloc((size_t)H * 512 * 2);
  bf16*  W2B     = (bf16*) alloc((size_t)H * H * 2);
  bf16*  Wo1B    = (bf16*) alloc((size_t)H * H * 2);
  bf16*  Wo2B    = (bf16*) alloc((size_t)H * H * 2);

  hipMemsetAsync(d_ws, 0, 8192, stream);

  // --- precision conversion / padding ---
  pad_bf16<<<(ROWS * 32 + 255) / 256, 256, 0, stream>>>(polylines, padA, ROWS, C, 32);
  pad_bf16<<<(H * 32 + 255) / 256, 256, 0, stream>>>(W_pre, WpreB, H, C, 32);
  cvt_bf16<<<(H * 512 + 255) / 256, 256, 0, stream>>>(W1, W1B, H * 512);
  cvt_bf16<<<(H * H + 255) / 256, 256, 0, stream>>>(W2, W2B, H * H);
  cvt_bf16<<<(H * H + 255) / 256, 256, 0, stream>>>(Wo1, Wo1B, H * H);
  cvt_bf16<<<(H * H + 255) / 256, 256, 0, stream>>>(Wo2, Wo2B, H * H);
  mask_kernel<<<ROWS / 256, 256, 0, stream>>>(mask, mrow, cnt, ROWS);
  valid_kernel<<<(POLY + 255) / 256, 256, 0, stream>>>(mask, validF, POLY);

  const int gBig = (ROWS / 64) * 4 / 8;   // 1920 blocks (8 waves each)
  const int gSml = (POLY / 64) * 4 / 8;   // 96 blocks

  // --- layer 1: pre-MLP GEMM (K padded 9->32) + BN + ReLU + maxpool ---
  gemm_wmma<0, false><<<gBig, 256, 0, stream>>>(padA, nullptr, WpreB, nullptr,
      nullptr, actF, nullptr, ROWS, H, 32, 32);
  bn_stats<<<240, 256, 0, stream>>>(actF, mrow, sum1, sq1, ROWS);
  bn_finalize<<<1, 256, 0, stream>>>(sum1, sq1, g_pre, b_pre, cnt, scale1, shift1);
  bn_apply<0><<<POLY, 256, 0, stream>>>(actF, mrow, scale1, shift1, actB, pooledB);

  // --- layer 2: concat(x,pool) GEMM (K=512) + BN + ReLU ---
  gemm_wmma<0, true><<<gBig, 256, 0, stream>>>(actB, pooledB, W1B, nullptr,
      nullptr, actF, nullptr, ROWS, H, 512, 256);
  bn_stats<<<240, 256, 0, stream>>>(actF, mrow, sum2, sq2, ROWS);
  bn_finalize<<<1, 256, 0, stream>>>(sum2, sq2, g1, b1, cnt, scale2, shift2);
  bn_apply<1><<<POLY, 256, 0, stream>>>(actF, mrow, scale2, shift2, actB, nullptr);

  // --- layer 3: GEMM (K=256) + BN + ReLU + maxpool -> buf ---
  gemm_wmma<0, false><<<gBig, 256, 0, stream>>>(actB, nullptr, W2B, nullptr,
      nullptr, actF, nullptr, ROWS, H, 256, 256);
  bn_stats<<<240, 256, 0, stream>>>(actF, mrow, sum3, sq3, ROWS);
  bn_finalize<<<1, 256, 0, stream>>>(sum3, sq3, g2, b2, cnt, scale3, shift3);
  bn_apply<2><<<POLY, 256, 0, stream>>>(actF, mrow, scale3, shift3, nullptr, bufB);

  // --- out MLP: relu(buf@Wo1^T+bo1) @ Wo2^T + bo2, masked by valid ---
  gemm_wmma<1, false><<<gSml, 256, 0, stream>>>(bufB, nullptr, Wo1B, bo1,
      nullptr, nullptr, o1B, POLY, H, 256, 256);
  gemm_wmma<2, false><<<gSml, 256, 0, stream>>>(o1B, nullptr, Wo2B, bo2,
      validF, out, nullptr, POLY, H, 256, 256);
}
